// DiT_20581483283113
// MI455X (gfx1250) — matrix operand order repaired and checked by reading, not compiled, not run on hardware
//
#include <hip/hip_runtime.h>

#define BB 2
#define NN 2304
#define CC 384
#define NHEADS 6
#define HDIM 64
#define HH 48
#define NA 144
#define NZ 576
#define HZ 24
#define ATT_SCALE 0.125f

typedef __attribute__((ext_vector_type(16))) _Float16 v16h;
typedef __attribute__((ext_vector_type(8)))  _Float16 v8h;
typedef __attribute__((ext_vector_type(8)))  float    v8f;

__device__ __forceinline__ v16h cat8(v8h a, v8h b) {
  return __builtin_shufflevector(a, b, 0,1,2,3,4,5,6,7,8,9,10,11,12,13,14,15);
}
// A-matrix 16x32 f16 fragment: lanes 0-15 hold K={k0+0..7, k0+16..23},
// lanes 16-31 hold K={k0+8..15, k0+24..31} for row M = lane%16 (ISA 7.12.2).
__device__ __forceinline__ v16h load_a_frag(const _Float16* row, int k0, int hi) {
  v8h a0 = *(const v8h*)(row + k0 + hi * 8);
  v8h a1 = *(const v8h*)(row + k0 + 16 + hi * 8);
  return cat8(a0, a1);
}
// B-matrix 32x16 f16 fragment: lanes 0-15 hold K={k0+0..7, k0+16..23},
// lanes 16-31 hold K={k0+8..15, k0+24..31}, column N = lane%16 (matches A).
__device__ __forceinline__ v16h load_b_frag(const _Float16* row, int k0, int hi) {
  const _Float16* p = row + k0 + hi * 8;
  return cat8(*(const v8h*)p, *(const v8h*)(p + 16));
}
__device__ __forceinline__ v8f wmma32(v16h a, v16h b, v8f c) {
  return __builtin_amdgcn_wmma_f32_16x16x32_f16(false, a, false, b, (short)0, c,
                                                false, false);
}

// ---------------- elementwise conversion ----------------
__global__ void k_cvt16(const float* __restrict__ src, _Float16* __restrict__ dst, int n) {
  int i = blockIdx.x * blockDim.x + threadIdx.x;
  if (i < n) dst[i] = (_Float16)src[i];
}

// ---------------- avg-pool 2x2 + layernorm -> f16 anchors ----------------
__global__ void __launch_bounds__(128) k_pool_ln(const float* __restrict__ z,
                                                 const float* __restrict__ g,
                                                 const float* __restrict__ beta,
                                                 _Float16* __restrict__ anc) {
  __shared__ float sbuf[CC];
  __shared__ float red[128], red2[128];
  int b = blockIdx.x / NA, a = blockIdx.x % NA;
  int i = a / 12, j = a % 12;
  const float* z00 = z + ((size_t)b * NZ + (2 * i) * HZ + 2 * j) * CC;
  const float* z01 = z00 + CC;
  const float* z10 = z00 + HZ * CC;
  const float* z11 = z10 + CC;
  int tid = threadIdx.x;
  float s = 0.f, s2 = 0.f;
  for (int c = tid; c < CC; c += 128) {
    float v = 0.25f * (z00[c] + z01[c] + z10[c] + z11[c]);
    sbuf[c] = v; s += v; s2 += v * v;
  }
  red[tid] = s; red2[tid] = s2;
  __syncthreads();
  for (int o = 64; o > 0; o >>= 1) {
    if (tid < o) { red[tid] += red[tid + o]; red2[tid] += red2[tid + o]; }
    __syncthreads();
  }
  float mean = red[0] / CC;
  float var = red2[0] / CC - mean * mean;
  float rstd = rsqrtf(var + 1e-5f);
  _Float16* outp = anc + ((size_t)b * NA + a) * CC;
  for (int c = tid; c < CC; c += 128)
    outp[c] = (_Float16)((sbuf[c] - mean) * rstd * g[c] + beta[c]);
}

// ---------------- WMMA GEMM: Y[M,Nout] = X[M,K](f16) * W[Nout,K]^T + bias ----
// mode: 1 = write f32, 2 = sigmoid, 4 = write f16. One wave -> 16x64 tile.
// k-loop unrolled x2 with ping-pong fragment buffers and consume-then-refill
// ordering: each WMMA's fragments were loaded a half-iteration earlier, so the
// backend emits partial loadcnt waits and no cross-buffer register copies.
// Requires Kd % 64 == 0 (holds: 384 and 1152).
__global__ void __launch_bounds__(256) k_gemm(const _Float16* __restrict__ X,
                                              const _Float16* __restrict__ W,
                                              const float* __restrict__ bias,
                                              float* of32, _Float16* of16,
                                              int M, int Nout, int Kd, int mode) {
  int wave = (int)((blockIdx.x * blockDim.x + threadIdx.x) >> 5);
  int lane = threadIdx.x & 31;
  int lm = lane & 15, hi = lane >> 4;
  int nb64 = Nout >> 6;
  if (wave >= (M >> 4) * nb64) return;
  int mt = wave / nb64, nb = wave % nb64;

  const _Float16* xrow = X + (size_t)(mt * 16 + lm) * Kd;
  const _Float16* wrow0 = W + (size_t)(nb * 64 + lm) * Kd;

  v8f acc[4];
#pragma unroll
  for (int t = 0; t < 4; t++)
#pragma unroll
    for (int r = 0; r < 8; r++) acc[t][r] = 0.f;

  v16h A0 = load_a_frag(xrow, 0, hi);
  v16h B0[4];
#pragma unroll
  for (int t = 0; t < 4; t++) B0[t] = load_b_frag(wrow0 + (size_t)t * 16 * Kd, 0, hi);
  v16h A1 = load_a_frag(xrow, 32, hi);
  v16h B1[4];
#pragma unroll
  for (int t = 0; t < 4; t++) B1[t] = load_b_frag(wrow0 + (size_t)t * 16 * Kd, 32, hi);

  for (int k0 = 64; k0 < Kd; k0 += 64) {
#pragma unroll
    for (int t = 0; t < 4; t++) acc[t] = wmma32(A0, B0[t], acc[t]);
    A0 = load_a_frag(xrow, k0, hi);
#pragma unroll
    for (int t = 0; t < 4; t++)
      B0[t] = load_b_frag(wrow0 + (size_t)t * 16 * Kd, k0, hi);
#pragma unroll
    for (int t = 0; t < 4; t++) acc[t] = wmma32(A1, B1[t], acc[t]);
    A1 = load_a_frag(xrow, k0 + 32, hi);
#pragma unroll
    for (int t = 0; t < 4; t++)
      B1[t] = load_b_frag(wrow0 + (size_t)t * 16 * Kd, k0 + 32, hi);
  }
#pragma unroll
  for (int t = 0; t < 4; t++) acc[t] = wmma32(A0, B0[t], acc[t]);
#pragma unroll
  for (int t = 0; t < 4; t++) acc[t] = wmma32(A1, B1[t], acc[t]);

#pragma unroll
  for (int t = 0; t < 4; t++) {
    int col = nb * 64 + t * 16 + lm;
    float bv = bias[col];
#pragma unroll
    for (int r = 0; r < 8; r++) {
      int row = mt * 16 + r + hi * 8;
      float v = acc[t][r] + bv;
      if (mode & 2) v = 1.f / (1.f + __expf(-v));
      size_t idx = (size_t)row * Nout + col;
      if (mode & 1) of32[idx] = v;
      if (mode & 4) of16[idx] = (_Float16)v;
    }
  }
}

// ---------------- local 7x7 neighborhood attention (one wave / 4x4 tile) ----
__global__ void __launch_bounds__(32) k_local_attn(const _Float16* __restrict__ q16,
                                                   const _Float16* __restrict__ k16,
                                                   const _Float16* __restrict__ v16,
                                                   float* __restrict__ oloc) {
  __shared__ _Float16 kpat[112][64];   // [key patch][hd]
  __shared__ _Float16 vT[64][128];     // [hd][key patch] (zero-padded keys)
  __shared__ _Float16 attn[16][128];   // [query][key patch] unnormalized probs

  int bid = blockIdx.x;
  int b = bid / (NHEADS * 144);
  int rem = bid % (NHEADS * 144);
  int head = rem / 144;
  int tile = rem % 144;
  int y0 = (tile / 12) * 4, x0 = (tile % 12) * 4;
  int lane = threadIdx.x;
  int lm = lane & 15, hi = lane >> 4;

  // stage the 10x10 key patch (zero fill outside image -> logit 0, value 0)
  for (int c = lane; c < 896; c += 32) {
    int p = c >> 3, h8 = c & 7;
    int ky = p / 10, kx = p - ky * 10;
    int gy = y0 - 3 + ky, gx = x0 - 3 + kx;
    v8h kv, vv;
#pragma unroll
    for (int e = 0; e < 8; e++) { kv[e] = (_Float16)0.f; vv[e] = (_Float16)0.f; }
    if (p < 100 && gy >= 0 && gy < HH && gx >= 0 && gx < HH) {
      size_t basei = ((size_t)b * NN + gy * HH + gx) * CC + head * HDIM + h8 * 8;
      kv = *(const v8h*)(k16 + basei);
      vv = *(const v8h*)(v16 + basei);
    }
    *(v8h*)&kpat[p][h8 * 8] = kv;
#pragma unroll
    for (int e = 0; e < 8; e++) vT[h8 * 8 + e][p] = vv[e];
  }
  for (int idx = lane; idx < 64 * 16; idx += 32)
    vT[idx >> 4][112 + (idx & 15)] = (_Float16)0.f;
  __syncthreads();

  // Q fragments (query m = lane%16 at (y0 + m/4, x0 + m%4))
  const _Float16* qrow =
      q16 + ((size_t)b * NN + (y0 + (lm >> 2)) * HH + (x0 + (lm & 3))) * CC + head * HDIM;
  v16h qa0 = load_a_frag(qrow, 0, hi);
  v16h qa1 = load_a_frag(qrow, 32, hi);

  v8f acc[7];
#pragma unroll
  for (int t = 0; t < 7; t++)
#pragma unroll
    for (int r = 0; r < 8; r++) acc[t][r] = 0.f;

#pragma unroll
  for (int t = 0; t < 7; t++) {
    const _Float16* krow = &kpat[t * 16 + lm][0];
    acc[t] = wmma32(qa0, load_b_frag(krow, 0, hi), acc[t]);
    acc[t] = wmma32(qa1, load_b_frag(krow, 32, hi), acc[t]);
  }

  // masked softmax, fully in registers (rows m = r + hi*8, col = t*16 + lane%16)
  float rmax[8], rsum[8];
#pragma unroll
  for (int r = 0; r < 8; r++) rmax[r] = -1e30f;
#pragma unroll
  for (int t = 0; t < 7; t++) {
    int p = t * 16 + lm;
    int ky = p / 10, kx = p - (p / 10) * 10;
    bool pv = p < 100;
#pragma unroll
    for (int r = 0; r < 8; r++) {
      int m = r + hi * 8;
      int dy = ky - (m >> 2), dx = kx - (m & 3);
      bool valid = pv && dy >= 0 && dy <= 6 && dx >= 0 && dx <= 6;
      float l = acc[t][r] * ATT_SCALE;
      acc[t][r] = l;
      if (valid) rmax[r] = fmaxf(rmax[r], l);
    }
  }
#pragma unroll
  for (int r = 0; r < 8; r++) {
    rmax[r] = fmaxf(rmax[r], __shfl_xor(rmax[r], 8, 32));
    rmax[r] = fmaxf(rmax[r], __shfl_xor(rmax[r], 4, 32));
    rmax[r] = fmaxf(rmax[r], __shfl_xor(rmax[r], 2, 32));
    rmax[r] = fmaxf(rmax[r], __shfl_xor(rmax[r], 1, 32));
    rsum[r] = 0.f;
  }
#pragma unroll
  for (int t = 0; t < 7; t++) {
    int p = t * 16 + lm;
    int ky = p / 10, kx = p - (p / 10) * 10;
    bool pv = p < 100;
#pragma unroll
    for (int r = 0; r < 8; r++) {
      int m = r + hi * 8;
      int dy = ky - (m >> 2), dx = kx - (m & 3);
      bool valid = pv && dy >= 0 && dy <= 6 && dx >= 0 && dx <= 6;
      float pe = valid ? __expf(acc[t][r] - rmax[r]) : 0.f;
      rsum[r] += pe;
      attn[m][p] = (_Float16)pe;
    }
  }
#pragma unroll
  for (int r = 0; r < 8; r++) {
    attn[r + hi * 8][112 + lm] = (_Float16)0.f;
    rsum[r] += __shfl_xor(rsum[r], 8, 32);
    rsum[r] += __shfl_xor(rsum[r], 4, 32);
    rsum[r] += __shfl_xor(rsum[r], 2, 32);
    rsum[r] += __shfl_xor(rsum[r], 1, 32);
  }
  __syncthreads();

  // out = attn * V  (16x128 * 128x64)
  v8f oacc[4];
#pragma unroll
  for (int t = 0; t < 4; t++)
#pragma unroll
    for (int r = 0; r < 8; r++) oacc[t][r] = 0.f;
  const _Float16* arow = &attn[lm][0];
#pragma unroll
  for (int ks = 0; ks < 4; ks++) {
    v16h af = load_a_frag(arow, ks * 32, hi);
#pragma unroll
    for (int nt = 0; nt < 4; nt++)
      oacc[nt] = wmma32(af, load_b_frag(&vT[nt * 16 + lm][0], ks * 32, hi), oacc[nt]);
  }
#pragma unroll
  for (int nt = 0; nt < 4; nt++)
#pragma unroll
    for (int r = 0; r < 8; r++) {
      int m = r + hi * 8;
      size_t row = (size_t)b * NN + (y0 + (m >> 2)) * HH + (x0 + (m & 3));
      oloc[row * CC + head * HDIM + nt * 16 + lm] = oacc[nt][r] / rsum[r];
    }
}

// ---------------- anchor cross-attention (144 keys = 9 full tiles) ----------
__global__ void __launch_bounds__(256) k_anchor_attn(const _Float16* __restrict__ q16,
                                                     const _Float16* __restrict__ ka,
                                                     const _Float16* __restrict__ va,
                                                     float* __restrict__ oanc) {
  __shared__ _Float16 vT[64][160];         // [hd][key] zero-padded to 160
  __shared__ _Float16 attn[8][16][160];    // per-wave probs

  int bid = blockIdx.x;
  int b = bid / (NHEADS * 18);
  int rem = bid % (NHEADS * 18);
  int head = rem / 18;
  int blk = rem % 18;
  int tid = threadIdx.x;
  int wv = tid >> 5, lane = tid & 31;
  int lm = lane & 15, hi = lane >> 4;

  for (int c = tid; c < NA * 8; c += 256) {
    int p = c >> 3, h8 = c & 7;
    v8h vv = *(const v8h*)(va + ((size_t)b * NA + p) * CC + head * HDIM + h8 * 8);
#pragma unroll
    for (int e = 0; e < 8; e++) vT[h8 * 8 + e][p] = vv[e];
  }
  for (int idx = tid; idx < 64 * 16; idx += 256)
    vT[idx >> 4][NA + (idx & 15)] = (_Float16)0.f;
  __syncthreads();

  int qbase = (blk * 8 + wv) * 16;
  const _Float16* qrow = q16 + ((size_t)b * NN + qbase + lm) * CC + head * HDIM;
  v16h qa0 = load_a_frag(qrow, 0, hi);
  v16h qa1 = load_a_frag(qrow, 32, hi);

  v8f acc[9];
#pragma unroll
  for (int t = 0; t < 9; t++)
#pragma unroll
    for (int r = 0; r < 8; r++) acc[t][r] = 0.f;

  // QK^T over 9 key tiles, prefetch tile t+1's B fragments before tile t's WMMAs
  const _Float16* krow0 = ka + ((size_t)b * NA + lm) * CC + head * HDIM;
  v16h kb0 = load_b_frag(krow0, 0, hi);
  v16h kb1 = load_b_frag(krow0, 32, hi);
#pragma unroll
  for (int t = 0; t < 9; t++) {
    v16h nb0, nb1;
    if (t < 8) {
      const _Float16* krow = krow0 + (size_t)(t + 1) * 16 * CC;
      nb0 = load_b_frag(krow, 0, hi);
      nb1 = load_b_frag(krow, 32, hi);
    }
    acc[t] = wmma32(qa0, kb0, acc[t]);
    acc[t] = wmma32(qa1, kb1, acc[t]);
    if (t < 8) { kb0 = nb0; kb1 = nb1; }
  }

  float rmax[8], rsum[8];
#pragma unroll
  for (int r = 0; r < 8; r++) rmax[r] = -1e30f;
#pragma unroll
  for (int t = 0; t < 9; t++)
#pragma unroll
    for (int r = 0; r < 8; r++) {
      float l = acc[t][r] * ATT_SCALE;
      acc[t][r] = l;
      rmax[r] = fmaxf(rmax[r], l);
    }
#pragma unroll
  for (int r = 0; r < 8; r++) {
    rmax[r] = fmaxf(rmax[r], __shfl_xor(rmax[r], 8, 32));
    rmax[r] = fmaxf(rmax[r], __shfl_xor(rmax[r], 4, 32));
    rmax[r] = fmaxf(rmax[r], __shfl_xor(rmax[r], 2, 32));
    rmax[r] = fmaxf(rmax[r], __shfl_xor(rmax[r], 1, 32));
    rsum[r] = 0.f;
  }
#pragma unroll
  for (int t = 0; t < 9; t++) {
    int p = t * 16 + lm;
#pragma unroll
    for (int r = 0; r < 8; r++) {
      float pe = __expf(acc[t][r] - rmax[r]);
      rsum[r] += pe;
      attn[wv][r + hi * 8][p] = (_Float16)pe;
    }
  }
#pragma unroll
  for (int r = 0; r < 8; r++) {
    attn[wv][r + hi * 8][NA + lm] = (_Float16)0.f;
    rsum[r] += __shfl_xor(rsum[r], 8, 32);
    rsum[r] += __shfl_xor(rsum[r], 4, 32);
    rsum[r] += __shfl_xor(rsum[r], 2, 32);
    rsum[r] += __shfl_xor(rsum[r], 1, 32);
  }
  __syncthreads();

  v8f oacc[4];
#pragma unroll
  for (int t = 0; t < 4; t++)
#pragma unroll
    for (int r = 0; r < 8; r++) oacc[t][r] = 0.f;
  const _Float16* arow = &attn[wv][lm][0];
#pragma unroll
  for (int ks = 0; ks < 5; ks++) {
    v16h af = load_a_frag(arow, ks * 32, hi);
#pragma unroll
    for (int nt = 0; nt < 4; nt++)
      oacc[nt] = wmma32(af, load_b_frag(&vT[nt * 16 + lm][0], ks * 32, hi), oacc[nt]);
  }
#pragma unroll
  for (int nt = 0; nt < 4; nt++)
#pragma unroll
    for (int r = 0; r < 8; r++) {
      int m = r + hi * 8;
      size_t row = (size_t)b * NN + qbase + m;
      oanc[row * CC + head * HDIM + nt * 16 + lm] = oacc[nt][r] / rsum[r];
    }
}

// ---------------- concat [x, o_loc, o_anc] -> f16 ----------------
__global__ void k_concat(const float* __restrict__ x, const float* __restrict__ ol,
                         const float* __restrict__ oa, _Float16* __restrict__ cat) {
  int i = blockIdx.x * 256 + threadIdx.x;
  if (i >= BB * NN * CC) return;
  int row = i / CC, col = i - row * CC;
  _Float16* d = cat + (size_t)row * (3 * CC) + col;
  d[0] = (_Float16)x[i];
  d[CC] = (_Float16)ol[i];
  d[2 * CC] = (_Float16)oa[i];
}

// ---------------- out = o_loc + g * o_anc -> f16 ----------------
__global__ void k_gate(const float* __restrict__ ol, const float* __restrict__ oa,
                       const float* __restrict__ g, _Float16* __restrict__ o16) {
  int i = blockIdx.x * 256 + threadIdx.x;
  if (i >= BB * NN * CC) return;
  o16[i] = (_Float16)(ol[i] + g[i] * oa[i]);
}

extern "C" void kernel_launch(void* const* d_in, const int* in_sizes, int n_in,
                              void* d_out, int out_size, void* d_ws, size_t ws_size,
                              hipStream_t stream) {
  (void)in_sizes; (void)n_in; (void)out_size; (void)ws_size;
  const float* x      = (const float*)d_in[0];
  const float* z      = (const float*)d_in[1];
  const float* Wq_loc = (const float*)d_in[2];  const float* bq_loc = (const float*)d_in[3];
  const float* Wk_loc = (const float*)d_in[4];  const float* bk_loc = (const float*)d_in[5];
  const float* Wv_loc = (const float*)d_in[6];  const float* bv_loc = (const float*)d_in[7];
  const float* Wq_anc = (const float*)d_in[8];  const float* bq_anc = (const float*)d_in[9];
  const float* Wk_anc = (const float*)d_in[10]; const float* bk_anc = (const float*)d_in[11];
  const float* Wv_anc = (const float*)d_in[12]; const float* bv_anc = (const float*)d_in[13];
  const float* Wg     = (const float*)d_in[14]; const float* bg     = (const float*)d_in[15];
  const float* Wo     = (const float*)d_in[16]; const float* bo     = (const float*)d_in[17];
  const float* ln_g   = (const float*)d_in[18]; const float* ln_b   = (const float*)d_in[19];
  float* out = (float*)d_out;

  char* base = (char*)d_ws;
  size_t off = 0;
  auto alloc = [&](size_t bytes) -> void* {
    void* p = base + off;
    off += (bytes + 255) & ~(size_t)255;
    return p;
  };
  const size_t TOK = (size_t)BB * NN;  // 4608

  _Float16* x16   = (_Float16*)alloc(TOK * CC * 2);
  _Float16* wq_l  = (_Float16*)alloc((size_t)CC * CC * 2);
  _Float16* wk_l  = (_Float16*)alloc((size_t)CC * CC * 2);
  _Float16* wv_l  = (_Float16*)alloc((size_t)CC * CC * 2);
  _Float16* wq_a  = (_Float16*)alloc((size_t)CC * CC * 2);
  _Float16* wk_a  = (_Float16*)alloc((size_t)CC * CC * 2);
  _Float16* wv_a  = (_Float16*)alloc((size_t)CC * CC * 2);
  _Float16* wo16  = (_Float16*)alloc((size_t)CC * CC * 2);
  _Float16* wg16  = (_Float16*)alloc((size_t)CC * 3 * CC * 2);
  _Float16* anc16 = (_Float16*)alloc((size_t)BB * NA * CC * 2);
  _Float16* qloc  = (_Float16*)alloc(TOK * CC * 2);
  _Float16* kloc  = (_Float16*)alloc(TOK * CC * 2);
  _Float16* vloc  = (_Float16*)alloc(TOK * CC * 2);
  _Float16* qanc  = (_Float16*)alloc(TOK * CC * 2);
  _Float16* kanc  = (_Float16*)alloc((size_t)BB * NA * CC * 2);
  _Float16* vanc  = (_Float16*)alloc((size_t)BB * NA * CC * 2);
  float*    oloc  = (float*)alloc(TOK * CC * 4);
  float*    oanc  = (float*)alloc(TOK * CC * 4);
  _Float16* cat16 = (_Float16*)alloc(TOK * 3 * CC * 2);
  float*    gbuf  = (float*)alloc(TOK * CC * 4);
  _Float16* out16 = (_Float16*)alloc(TOK * CC * 2);

  auto cvt = [&](const float* s, _Float16* d, size_t n) {
    k_cvt16<<<(unsigned)((n + 255) / 256), 256, 0, stream>>>(s, d, (int)n);
  };
  cvt(x, x16, TOK * CC);
  cvt(Wq_loc, wq_l, (size_t)CC * CC);
  cvt(Wk_loc, wk_l, (size_t)CC * CC);
  cvt(Wv_loc, wv_l, (size_t)CC * CC);
  cvt(Wq_anc, wq_a, (size_t)CC * CC);
  cvt(Wk_anc, wk_a, (size_t)CC * CC);
  cvt(Wv_anc, wv_a, (size_t)CC * CC);
  cvt(Wo, wo16, (size_t)CC * CC);
  cvt(Wg, wg16, (size_t)CC * 3 * CC);

  k_pool_ln<<<BB * NA, 128, 0, stream>>>(z, ln_g, ln_b, anc16);

  auto gemm = [&](const _Float16* X, const _Float16* W, const float* bias,
                  float* of32, _Float16* of16, int M, int Nout, int Kd, int mode) {
    int waves = (M / 16) * (Nout / 64);
    k_gemm<<<(waves + 7) / 8, 256, 0, stream>>>(X, W, bias, of32, of16, M, Nout, Kd, mode);
  };

  gemm(x16, wq_l, bq_loc, nullptr, qloc, (int)TOK, CC, CC, 4);
  gemm(x16, wk_l, bk_loc, nullptr, kloc, (int)TOK, CC, CC, 4);
  gemm(x16, wv_l, bv_loc, nullptr, vloc, (int)TOK, CC, CC, 4);
  gemm(x16, wq_a, bq_anc, nullptr, qanc, (int)TOK, CC, CC, 4);
  gemm(anc16, wk_a, bk_anc, nullptr, kanc, BB * NA, CC, CC, 4);
  gemm(anc16, wv_a, bv_anc, nullptr, vanc, BB * NA, CC, CC, 4);

  k_local_attn<<<BB * NHEADS * 144, 32, 0, stream>>>(qloc, kloc, vloc, oloc);
  k_anchor_attn<<<BB * NHEADS * 18, 256, 0, stream>>>(qanc, kanc, vanc, oanc);

  k_concat<<<(unsigned)((TOK * CC + 255) / 256), 256, 0, stream>>>(x, oloc, oanc, cat16);
  gemm(cat16, wg16, bg, gbuf, nullptr, (int)TOK, CC, 3 * CC, 1 | 2);
  k_gate<<<(unsigned)((TOK * CC + 255) / 256), 256, 0, stream>>>(oloc, oanc, gbuf, out16);
  gemm(out16, wo16, bo, out, nullptr, (int)TOK, CC, CC, 1);
}
